// TransformerMoE_19980187861340
// MI455X (gfx1250) — matrix operand order repaired and checked by reading, not compiled, not run on hardware
//
#include <hip/hip_runtime.h>
#include <hip/hip_bf16.h>

// ---------------- model constants ----------------
#define BB   4
#define TT   1024
#define VV   32000
#define DD   1024
#define NHH  16
#define HSS  64
#define EE   8
#define HID  2816
#define CAP  1024
#define NTOK (BB*TT)
#define EPSF 1e-5f

// ---------------- types / helpers ----------------
typedef __attribute__((ext_vector_type(16))) __bf16 bf16x16;
typedef __attribute__((ext_vector_type(8)))  float  f32x8;
typedef __attribute__((ext_vector_type(4)))  unsigned int u32x4;
typedef __attribute__((ext_vector_type(8)))  int    i32x8;
typedef __attribute__((ext_vector_type(4)))  int    i32x4;

union Frag { bf16x16 v; uint4 q[2]; };

__device__ __forceinline__ unsigned short f2bf(float f) {
  unsigned u = __builtin_bit_cast(unsigned, f);
  u += 0x7FFFu + ((u >> 16) & 1u);           // round-to-nearest-even
  return (unsigned short)(u >> 16);
}
__device__ __forceinline__ float bf2f(unsigned short u) {
  return __builtin_bit_cast(float, (unsigned)u << 16);
}

// ---------------- Tensor Data Mover availability ----------------
#if __has_builtin(__builtin_amdgcn_tensor_load_to_lds) && \
    __has_builtin(__builtin_amdgcn_s_wait_tensorcnt)
#define USE_TDM 1
#else
#define USE_TDM 0
#endif

#if USE_TDM
// Issue one TDM 2D tile load: 32 elems/row x 128 rows of bf16 from a row-major
// [rows x K] matrix starting at gptr, into LDS at lds_off with an 80B padded
// row stride (pad_interval = 16 DWORDs, pad_amount = 4 DWORDs -> 64B+16B).
__device__ __forceinline__ void tdm_load_tile(unsigned lds_off,
                                              const unsigned short* gptr,
                                              unsigned rows, unsigned dim0,
                                              unsigned stride_elems) {
  unsigned long long ga = (unsigned long long)(size_t)gptr;
  u32x4 g0;
  g0[0] = 1u;                                        // count=1, user descriptor
  g0[1] = lds_off;                                   // lds_addr (bytes)
  g0[2] = (unsigned)(ga & 0xFFFFFFFFu);              // global_addr[31:0]
  g0[3] = (unsigned)((ga >> 32) & 0x01FFFFFFu)       // global_addr[56:32]
        | (2u << 30);                                // type = 2 ("image")
  i32x8 g1;
  g1[0] = (int)((1u << 16)                           // data_size = 2 bytes
              | (1u << 20)                           // pad_enable
              | (3u << 22)                           // pad_interval: 16 DWORDs
              | (3u << 25));                         // pad_amount : 4 DWORDs
  g1[1] = (int)((dim0 & 0xFFFFu) << 16);             // tensor_dim0[15:0]
  g1[2] = (int)((dim0 >> 16) | ((rows & 0xFFFFu) << 16)); // dim0 hi | dim1 lo
  g1[3] = (int)((rows >> 16) | (32u << 16));         // dim1 hi | tile_dim0=32
  g1[4] = (int)(128u);                               // tile_dim1=128, tile_dim2=0
  g1[5] = (int)stride_elems;                         // tensor_dim0_stride lo
  g1[6] = 0;                                         // stride hi | dim1_stride lo
  g1[7] = 0;
  i32x4 z4 = {0, 0, 0, 0};
#if __clang_major__ >= 23
  i32x8 z8 = {0, 0, 0, 0, 0, 0, 0, 0};
  __builtin_amdgcn_tensor_load_to_lds(g0, g1, z4, z4, z8, 0);
#else
  __builtin_amdgcn_tensor_load_to_lds(g0, g1, z4, z4, 0);
#endif
}
#endif  // USE_TDM

// =====================================================================
// Generic NT GEMM: C[M,N](f32) = A[M,K](bf16,row) * B[N,K](bf16,row)^T (+bias[N])
// 128x128x32 tile, 256 thr = 8 waves (2M x 4N), each wave 64x32 = 8 WMMA accs.
// TDM path: double-buffered LDS, wave0 issues tensor_load_to_lds for stage
// k+2 after stage k is consumed, while all waves WMMA stage k+1; waits pipelined via s_wait_tensorcnt.
// =====================================================================
#define TILE_HALVES (128 * 40)
#define TILE_BYTES  (TILE_HALVES * 2)

__global__ __launch_bounds__(256)
void k_gemm_nt(const unsigned short* __restrict__ A,
               const unsigned short* __restrict__ B,
               float* __restrict__ C, const float* __restrict__ bias,
               int M, int N, int K,
               long long strideA, long long strideB, long long strideC)
{
#if USE_TDM
  __shared__ unsigned short sA[2][TILE_HALVES];
  __shared__ unsigned short sB[2][TILE_HALVES];
#else
  __shared__ unsigned short sA[1][TILE_HALVES];
  __shared__ unsigned short sB[1][TILE_HALVES];
#endif

  const int tid  = threadIdx.x;
  const int lane = tid & 31;
  const int wv   = tid >> 5;
  const int wm   = wv >> 2;                 // 0..1
  const int wn   = wv & 3;                  // 0..3
  const int bm   = blockIdx.y * 128;
  const int bn   = blockIdx.x * 128;

  A += (size_t)blockIdx.z * strideA;
  B += (size_t)blockIdx.z * strideB;
  C += (size_t)blockIdx.z * strideC;

  f32x8 acc[4][2] = {};

  const int r    = lane & 15;
  const int ka   = (lane < 16) ? 0 : 8;     // A frag: K chunks k..k+7, k+16..k+23
  const int kb   = (lane < 16) ? 0 : 8;     // B frag: K chunks k..k+7, k+16..k+23
  const int nk   = K >> 5;                  // K assumed multiple of 32

#if USE_TDM
  const unsigned ldsA = (unsigned)(size_t)(void*)&sA[0][0];
  const unsigned ldsB = (unsigned)(size_t)(void*)&sB[0][0];
  // ---- prologue: put stages 0 and 1 in flight ----
  if (wv == 0) {
    tdm_load_tile(ldsA, A + (size_t)bm * K, (unsigned)(M - bm), (unsigned)K, (unsigned)K);
    tdm_load_tile(ldsB, B + (size_t)bn * K, (unsigned)(N - bn), (unsigned)K, (unsigned)K);
    if (nk > 1) {
      tdm_load_tile(ldsA + TILE_BYTES, A + (size_t)bm * K + 32,
                    (unsigned)(M - bm), (unsigned)(K - 32), (unsigned)K);
      tdm_load_tile(ldsB + TILE_BYTES, B + (size_t)bn * K + 32,
                    (unsigned)(N - bn), (unsigned)(K - 32), (unsigned)K);
    }
  }
  for (int ki = 0; ki < nk; ++ki) {
    // stage ki done when at most the next stage's 2 ops remain outstanding
    if (ki + 1 < nk) __builtin_amdgcn_s_wait_tensorcnt(2);
    else             __builtin_amdgcn_s_wait_tensorcnt(0);
    __syncthreads();
    const int buf = ki & 1;

    Frag a[4], b[2];
    #pragma unroll
    for (int i = 0; i < 4; ++i) {
      const unsigned short* p = &sA[buf][(wm * 64 + i * 16 + r) * 40 + ka];
      a[i].q[0] = *(const uint4*)p;
      a[i].q[1] = *(const uint4*)(p + 16);
    }
    #pragma unroll
    for (int j = 0; j < 2; ++j) {
      const unsigned short* p = &sB[buf][(wn * 32 + j * 16 + r) * 40 + kb];
      b[j].q[0] = *(const uint4*)p;
      b[j].q[1] = *(const uint4*)(p + 16);
    }
    #pragma unroll
    for (int i = 0; i < 4; ++i)
      #pragma unroll
      for (int j = 0; j < 2; ++j)
        acc[i][j] = __builtin_amdgcn_wmma_f32_16x16x32_bf16(
            false, a[i].v, false, b[j].v, (short)0, acc[i][j], false, false);
    __syncthreads();

    if (ki + 2 < nk && wv == 0) {   // refill the buffer we just consumed
      const int kk = (ki + 2) * 32;
      tdm_load_tile(ldsA + (unsigned)buf * TILE_BYTES, A + (size_t)bm * K + kk,
                    (unsigned)(M - bm), (unsigned)(K - kk), (unsigned)K);
      tdm_load_tile(ldsB + (unsigned)buf * TILE_BYTES, B + (size_t)bn * K + kk,
                    (unsigned)(N - bn), (unsigned)(K - kk), (unsigned)K);
    }
  }
#else
  // ---- fallback: manual staging through VGPRs ----
  const int lrow = tid >> 1;                // 0..127
  const int lcol = (tid & 1) * 16;          // 0 | 16 (halves)
  for (int ki = 0; ki < nk; ++ki) {
    const int k0 = ki * 32;
    {
      uint4 z; z.x = z.y = z.z = z.w = 0u;
      uint4 a0 = z, a1 = z, b0 = z, b1 = z;
      int ga = bm + lrow, gb = bn + lrow;
      if (ga < M) {
        const uint4* p = (const uint4*)(A + (size_t)ga * K + k0 + lcol);
        a0 = p[0]; a1 = p[1];
      }
      if (gb < N) {
        const uint4* p = (const uint4*)(B + (size_t)gb * K + k0 + lcol);
        b0 = p[0]; b1 = p[1];
      }
      *(uint4*)(sA[0] + lrow * 40 + lcol)     = a0;
      *(uint4*)(sA[0] + lrow * 40 + lcol + 8) = a1;
      *(uint4*)(sB[0] + lrow * 40 + lcol)     = b0;
      *(uint4*)(sB[0] + lrow * 40 + lcol + 8) = b1;
    }
    __syncthreads();
    Frag a[4], b[2];
    #pragma unroll
    for (int i = 0; i < 4; ++i) {
      const unsigned short* p = &sA[0][(wm * 64 + i * 16 + r) * 40 + ka];
      a[i].q[0] = *(const uint4*)p;
      a[i].q[1] = *(const uint4*)(p + 16);
    }
    #pragma unroll
    for (int j = 0; j < 2; ++j) {
      const unsigned short* p = &sB[0][(wn * 32 + j * 16 + r) * 40 + kb];
      b[j].q[0] = *(const uint4*)p;
      b[j].q[1] = *(const uint4*)(p + 16);
    }
    #pragma unroll
    for (int i = 0; i < 4; ++i)
      #pragma unroll
      for (int j = 0; j < 2; ++j)
        acc[i][j] = __builtin_amdgcn_wmma_f32_16x16x32_bf16(
            false, a[i].v, false, b[j].v, (short)0, acc[i][j], false, false);
    __syncthreads();
  }
#endif

  // ---- epilogue: C layout = VGPR j -> M = j + (lane>=16 ? 8 : 0), N = lane&15
  const int mh = (lane >= 16) ? 8 : 0;
  #pragma unroll
  for (int i = 0; i < 4; ++i)
    #pragma unroll
    for (int j = 0; j < 2; ++j) {
      int col = bn + wn * 32 + j * 16 + (lane & 15);
      if (col >= N) continue;
      float badd = bias ? bias[col] : 0.f;
      #pragma unroll
      for (int e = 0; e < 8; ++e) {
        int row = bm + wm * 64 + i * 16 + e + mh;
        if (row < M) C[(size_t)row * N + col] = acc[i][j][e] + badd;
      }
    }
}

// =====================================================================
// Fused causal attention (flash-style), bf16 WMMA, fp32 online softmax.
// grid=(T/64, B*NH), block=128 (4 waves); each wave owns 16 query rows.
// Q,K stored [b,h,t,e]; V stored transposed [b,h,e,t] so all WMMA operand
// loads are two runs of 8 contiguous bf16 per lane. P re-layout via 1KB/wave LDS.
// =====================================================================
__global__ __launch_bounds__(128)
void k_attn(const unsigned short* __restrict__ qbf,
            const unsigned short* __restrict__ kbf,
            const unsigned short* __restrict__ vtb,
            unsigned short* __restrict__ abf)
{
  __shared__ unsigned short P[4][16 * 32];

  const int lane = threadIdx.x & 31;
  const int wv   = threadIdx.x >> 5;
  const int bh   = blockIdx.y;
  const int b    = bh / NHH, h = bh % NHH;
  const unsigned short* Q  = qbf + (size_t)bh * TT * HSS;
  const unsigned short* Km = kbf + (size_t)bh * TT * HSS;
  const unsigned short* Vt = vtb + (size_t)bh * HSS * TT;
  const int q0 = blockIdx.x * 64 + wv * 16;

  const int r  = lane & 15;
  const int ka = (lane < 16) ? 0 : 8;
  const int kb = (lane < 16) ? 0 : 8;
  const int mh = (lane >= 16) ? 8 : 0;

  Frag aq[2];
  #pragma unroll
  for (int c = 0; c < 2; ++c) {
    const unsigned short* p = Q + (size_t)(q0 + r) * HSS + c * 32 + ka;
    aq[c].q[0] = *(const uint4*)p;
    aq[c].q[1] = *(const uint4*)(p + 16);
  }

  float rm[8], rl[8];
  #pragma unroll
  for (int j = 0; j < 8; ++j) { rm[j] = -3.0e38f; rl[j] = 0.f; }
  f32x8 o[4] = {};

  for (int s0 = 0; s0 < q0 + 16; s0 += 32) {
    // ---- S = Q K^T for a 16x32 slab ----
    f32x8 sc[2] = {};
    #pragma unroll
    for (int n = 0; n < 2; ++n) {
      const int scol = s0 + n * 16 + r;
      #pragma unroll
      for (int c = 0; c < 2; ++c) {
        Frag kf;
        const unsigned short* p = Km + (size_t)scol * HSS + c * 32 + kb;
        kf.q[0] = *(const uint4*)p;
        kf.q[1] = *(const uint4*)(p + 16);
        sc[n] = __builtin_amdgcn_wmma_f32_16x16x32_bf16(
            false, aq[c].v, false, kf.v, (short)0, sc[n], false, false);
      }
    }
    // ---- causal mask + online softmax (row = VGPR idx, 16 lanes = cols) ----
    float pv0[8], pv1[8];
    #pragma unroll
    for (int j = 0; j < 8; ++j) {
      const int trow = q0 + j + mh;
      float v0 = sc[0][j] * 0.125f;               // 1/sqrt(64)
      float v1 = sc[1][j] * 0.125f;
      if (s0      + (lane & 15) > trow) v0 = -3.0e38f;
      if (s0 + 16 + (lane & 15) > trow) v1 = -3.0e38f;
      float mx = fmaxf(v0, v1);
      #pragma unroll
      for (int m = 8; m >= 1; m >>= 1) mx = fmaxf(mx, __shfl_xor(mx, m, 32));
      float mnew = fmaxf(rm[j], mx);
      float corr = __expf(rm[j] - mnew);
      float p0 = __expf(v0 - mnew), p1 = __expf(v1 - mnew);
      float rs = p0 + p1;
      #pragma unroll
      for (int m = 8; m >= 1; m >>= 1) rs += __shfl_xor(rs, m, 32);
      rl[j] = rl[j] * corr + rs;
      rm[j] = mnew;
      #pragma unroll
      for (int t = 0; t < 4; ++t) o[t][j] *= corr;
      pv0[j] = p0; pv1[j] = p1;
    }
    // ---- P -> per-wave LDS (C layout) -> reload as A fragment ----
    #pragma unroll
    for (int j = 0; j < 8; ++j) {
      P[wv][(j + mh) * 32 +      (lane & 15)] = f2bf(pv0[j]);
      P[wv][(j + mh) * 32 + 16 + (lane & 15)] = f2bf(pv1[j]);
    }
    asm volatile("s_wait_dscnt 0" ::: "memory"); // per-wave RAW through LDS
    Frag pa;
    {
      const unsigned short* p = &P[wv][r * 32 + ka];
      pa.q[0] = *(const uint4*)p;
      pa.q[1] = *(const uint4*)(p + 16);
    }
    // ---- O += P * V  (V transposed: lane = output feature, K contiguous) ----
    #pragma unroll
    for (int t = 0; t < 4; ++t) {
      Frag vf;
      const unsigned short* p = Vt + (size_t)(t * 16 + r) * TT + s0 + kb;
      vf.q[0] = *(const uint4*)p;
      vf.q[1] = *(const uint4*)(p + 16);
      o[t] = __builtin_amdgcn_wmma_f32_16x16x32_bf16(
          false, pa.v, false, vf.v, (short)0, o[t], false, false);
    }
  }

  // ---- normalize + store [b, t, h*64+e] as bf16 for the WO GEMM ----
  #pragma unroll
  for (int j = 0; j < 8; ++j) {
    const int trow = q0 + j + mh;
    const float inv = 1.f / rl[j];
    #pragma unroll
    for (int t = 0; t < 4; ++t) {
      int feat = h * HSS + t * 16 + (lane & 15);
      abf[((size_t)b * TT + trow) * DD + feat] = f2bf(o[t][j] * inv);
    }
  }
}

// =====================================================================
// Elementwise / small kernels
// =====================================================================
__global__ void k_embed(const int* __restrict__ idx, const float* __restrict__ emb,
                        float* __restrict__ x) {
  int bt = blockIdx.x;
  int tok = idx[bt];
  for (int d = threadIdx.x; d < DD; d += blockDim.x)
    x[(size_t)bt * DD + d] = emb[(size_t)tok * DD + d];
}

__global__ void k_rmsnorm(const float* __restrict__ x, const float* __restrict__ w,
                          unsigned short* __restrict__ out) {
  __shared__ float red[8];
  int row = blockIdx.x;
  float s = 0.f;
  for (int d = threadIdx.x; d < DD; d += 256) {
    float v = x[(size_t)row * DD + d]; s += v * v;
  }
  #pragma unroll
  for (int m = 16; m >= 1; m >>= 1) s += __shfl_xor(s, m, 32);
  if ((threadIdx.x & 31) == 0) red[threadIdx.x >> 5] = s;
  __syncthreads();
  float tot = 0.f;
  #pragma unroll
  for (int i = 0; i < 8; ++i) tot += red[i];
  float inv = rsqrtf(tot / (float)DD + EPSF);
  for (int d = threadIdx.x; d < DD; d += 256)
    out[(size_t)row * DD + d] = f2bf(x[(size_t)row * DD + d] * inv * w[d]);
}

__global__ void k_f2bf(const float* __restrict__ src, unsigned short* __restrict__ dst,
                       long long n) {
  long long i = (long long)blockIdx.x * blockDim.x + threadIdx.x;
  long long st = (long long)gridDim.x * blockDim.x;
  for (; i < n; i += st) dst[i] = f2bf(src[i]);
}

__global__ void k_add(float* __restrict__ x, const float* __restrict__ c, long long n) {
  long long i = (long long)blockIdx.x * blockDim.x + threadIdx.x;
  long long st = (long long)gridDim.x * blockDim.x;
  for (; i < n; i += st) x[i] += c[i];
}

__global__ void k_silu_mul(const float* __restrict__ c1, const float* __restrict__ c3,
                           unsigned short* __restrict__ h, long long n) {
  long long i = (long long)blockIdx.x * blockDim.x + threadIdx.x;
  long long st = (long long)gridDim.x * blockDim.x;
  for (; i < n; i += st) {
    float a = c1[i];
    h[i] = f2bf(a / (1.f + __expf(-a)) * c3[i]);
  }
}

// RoPE for q,k (even/odd split-rotate per the reference) + V transpose to [b,h,e,t]
__global__ void k_rope(const float* __restrict__ qa, const float* __restrict__ kA,
                       const float* __restrict__ va,
                       unsigned short* __restrict__ qbf, unsigned short* __restrict__ kbf,
                       unsigned short* __restrict__ vtb) {
  int bt = blockIdx.x;
  int t = bt % TT, b = bt / TT;
  for (int p = threadIdx.x; p < NHH * 32; p += blockDim.x) {
    int h = p >> 5, i = p & 31;
    float inv = __powf(10000.f, -2.f * (float)i / (float)HSS);
    float ang = (float)t * inv;
    float c = __cosf(ang), s = __sinf(ang);
    size_t ib = (size_t)bt * DD + h * HSS;
    size_t ob = ((size_t)(b * NHH + h) * TT + t) * HSS;
    float q1 = qa[ib + 2 * i], q2 = qa[ib + 2 * i + 1];
    qbf[ob + i]      = f2bf(q1 * c - q2 * s);
    qbf[ob + 32 + i] = f2bf(q1 * s + q2 * c);
    float k1 = kA[ib + 2 * i], k2 = kA[ib + 2 * i + 1];
    kbf[ob + i]      = f2bf(k1 * c - k2 * s);
    kbf[ob + 32 + i] = f2bf(k1 * s + k2 * c);
  }
  for (int f = threadIdx.x; f < DD; f += blockDim.x) {
    int h = f >> 6, e = f & 63;
    vtb[((size_t)(b * NHH + h) * HSS + e) * TT + t] = f2bf(va[(size_t)bt * DD + f]);
  }
}

// Router: one wave per token; 8 expert dots + top-2 + softmax gates.
__global__ void k_router(const unsigned short* __restrict__ xn,
                         const float* __restrict__ wr, const float* __restrict__ br,
                         int* __restrict__ topi, float* __restrict__ gates) {
  int tok = blockIdx.x, lane = threadIdx.x;
  float acc[EE] = {};
  for (int k = lane; k < DD; k += 32) {
    float xv = bf2f(xn[(size_t)tok * DD + k]);
    #pragma unroll
    for (int e = 0; e < EE; ++e) acc[e] += xv * wr[(size_t)e * DD + k];
  }
  #pragma unroll
  for (int e = 0; e < EE; ++e) {
    #pragma unroll
    for (int m = 16; m >= 1; m >>= 1) acc[e] += __shfl_xor(acc[e], m, 32);
    acc[e] += br[e];
  }
  if (lane == 0) {
    int i0 = 0; float v0 = acc[0];
    for (int e = 1; e < EE; ++e) if (acc[e] > v0) { v0 = acc[e]; i0 = e; }
    int i1 = (i0 == 0) ? 1 : 0; float v1 = acc[i1];
    for (int e = 0; e < EE; ++e)
      if (e != i0 && acc[e] > v1) { v1 = acc[e]; i1 = e; }
    float g1 = __expf(v1 - v0);
    float s = 1.f + g1;
    topi[tok * 2] = i0;  topi[tok * 2 + 1] = i1;
    gates[tok * 2] = 1.f / s; gates[tok * 2 + 1] = g1 / s;
  }
}

// Capacity assignment: one wave per expert, in-order ballot/popc scan of tokens.
__global__ void k_capacity(const int* __restrict__ topi, const float* __restrict__ gates,
                           int* __restrict__ slot, float* __restrict__ sgate,
                           int* __restrict__ cnt) {
  int e = blockIdx.x, lane = threadIdx.x;
  int base = 0;
  for (int t0 = 0; t0 < NTOK; t0 += 32) {
    int tok = t0 + lane;
    int i0 = topi[tok * 2], i1 = topi[tok * 2 + 1];
    bool sel = (i0 == e) || (i1 == e);
    float g = (i0 == e) ? gates[tok * 2] : gates[tok * 2 + 1];
    unsigned mask = (unsigned)__ballot(sel);
    int pref = __popc(mask & ((1u << lane) - 1u));
    if (sel) {
      int p = base + pref;
      if (p < CAP) { slot[e * CAP + p] = tok; sgate[e * CAP + p] = g; }
    }
    base += __popc(mask);
  }
  if (lane == 0) cnt[e] = (base < CAP) ? base : CAP;
}

__global__ void k_gather(const unsigned short* __restrict__ xn, const int* __restrict__ slot,
                         const int* __restrict__ cnt, unsigned short* __restrict__ xg) {
  int e = blockIdx.y, i = blockIdx.x;
  bool valid = i < cnt[e];
  int tok = valid ? slot[e * CAP + i] : 0;
  for (int d = threadIdx.x; d < DD; d += blockDim.x)
    xg[((size_t)e * CAP + i) * DD + d] = valid ? xn[(size_t)tok * DD + d] : (unsigned short)0;
}

__global__ void k_scatter(const float* __restrict__ yo, const int* __restrict__ slot,
                          const float* __restrict__ sgate, const int* __restrict__ cnt,
                          float* __restrict__ x) {
  int e = blockIdx.y, i = blockIdx.x;
  if (i >= cnt[e]) return;
  int tok = slot[e * CAP + i];
  float g = sgate[e * CAP + i];
  for (int d = threadIdx.x; d < DD; d += blockDim.x)
    atomicAdd(&x[(size_t)tok * DD + d], yo[((size_t)e * CAP + i) * DD + d] * g);
}

// =====================================================================
// Host-side orchestration
// =====================================================================
extern "C" void kernel_launch(void* const* d_in, const int* in_sizes, int n_in,
                              void* d_out, int out_size, void* d_ws, size_t ws_size,
                              hipStream_t stream) {
  (void)in_sizes; (void)n_in; (void)out_size; (void)ws_size;

  const int*   idx    = (const int*)d_in[0];
  const float* emb    = (const float*)d_in[1];
  const float* finalw = (const float*)d_in[2];
  const float* lnf    = (const float*)d_in[3];
  // per-layer field order: wq,bq,wk,bk,wv,bv,wo,bo,ln1,ln2,wr,br,wn,bn,w1,w3,w2
  const int LBASE[2] = {4, 21};

  // ---- workspace arena ----
  char* ws = (char*)d_ws;
  size_t off = 0;
  auto alloc = [&](size_t bytes) -> void* {
    off = (off + 255) & ~(size_t)255;
    void* p = ws + off;
    off += bytes;
    return p;
  };
  float*          x     = (float*)alloc((size_t)NTOK * DD * 4);
  unsigned short* xn    = (unsigned short*)alloc((size_t)NTOK * DD * 2);
  float*          qall  = (float*)alloc((size_t)NTOK * DD * 4);
  float*          kall  = (float*)alloc((size_t)NTOK * DD * 4);
  float*          vall  = (float*)alloc((size_t)NTOK * DD * 4);
  unsigned short* qbf   = (unsigned short*)alloc((size_t)NTOK * DD * 2);
  unsigned short* kbf   = (unsigned short*)alloc((size_t)NTOK * DD * 2);
  unsigned short* vtb   = (unsigned short*)alloc((size_t)NTOK * DD * 2);
  unsigned short* abf   = (unsigned short*)alloc((size_t)NTOK * DD * 2);
  float*          cattn = (float*)alloc((size_t)NTOK * DD * 4);
  unsigned short* wb    = (unsigned short*)alloc((size_t)VV * DD * 2);   // weight staging (max: final_w)
  int*            topi  = (int*)alloc((size_t)NTOK * 2 * 4);
  float*          gatev = (float*)alloc((size_t)NTOK * 2 * 4);
  int*            slot  = (int*)alloc((size_t)EE * CAP * 4);
  float*          sgate = (float*)alloc((size_t)EE * CAP * 4);
  int*            cnt   = (int*)alloc((size_t)EE * 4);
  unsigned short* xg    = (unsigned short*)alloc((size_t)EE * CAP * DD * 2);
  float*          c1    = (float*)alloc((size_t)EE * CAP * HID * 4);
  float*          c3    = (float*)alloc((size_t)EE * CAP * HID * 4);
  unsigned short* hb    = (unsigned short*)alloc((size_t)EE * CAP * HID * 2);
  float*          yo    = (float*)alloc((size_t)EE * CAP * DD * 4);

  auto conv = [&](const float* src, unsigned short* dst, long long n) {
    int blocks = (int)((n + 4095) / 4096);
    k_f2bf<<<blocks, 256, 0, stream>>>(src, dst, n);
  };
  auto gemm = [&](const unsigned short* A, const unsigned short* Bm, float* Cm,
                  const float* bias, int M, int N, int K, int batch,
                  long long sA, long long sB, long long sC) {
    dim3 g((N + 127) / 128, (M + 127) / 128, batch);
    k_gemm_nt<<<g, 256, 0, stream>>>(A, Bm, Cm, bias, M, N, K, sA, sB, sC);
  };

  // ---- embedding ----
  k_embed<<<NTOK, 256, 0, stream>>>(idx, emb, x);

  for (int layer = 0; layer < 2; ++layer) {
    const int Lb = LBASE[layer];
    const float* wq  = (const float*)d_in[Lb + 0];
    const float* bq  = (const float*)d_in[Lb + 1];
    const float* wk  = (const float*)d_in[Lb + 2];
    const float* bk  = (const float*)d_in[Lb + 3];
    const float* wv  = (const float*)d_in[Lb + 4];
    const float* bv  = (const float*)d_in[Lb + 5];
    const float* wo  = (const float*)d_in[Lb + 6];
    const float* bo  = (const float*)d_in[Lb + 7];
    const float* ln1 = (const float*)d_in[Lb + 8];
    const float* ln2 = (const float*)d_in[Lb + 9];
    const float* wr  = (const float*)d_in[Lb + 10];
    const float* br  = (const float*)d_in[Lb + 11];
    const float* w1  = (const float*)d_in[Lb + 14];
    const float* w3  = (const float*)d_in[Lb + 15];
    const float* w2  = (const float*)d_in[Lb + 16];

    // ---- attention block ----
    k_rmsnorm<<<NTOK, 256, 0, stream>>>(x, ln1, xn);
    conv(wq, wb, (long long)DD * DD);
    gemm(xn, wb, qall, bq, NTOK, DD, DD, 1, 0, 0, 0);
    conv(wk, wb, (long long)DD * DD);
    gemm(xn, wb, kall, bk, NTOK, DD, DD, 1, 0, 0, 0);
    conv(wv, wb, (long long)DD * DD);
    gemm(xn, wb, vall, bv, NTOK, DD, DD, 1, 0, 0, 0);
    k_rope<<<NTOK, 256, 0, stream>>>(qall, kall, vall, qbf, kbf, vtb);
    k_attn<<<dim3(TT / 64, BB * NHH), 128, 0, stream>>>(qbf, kbf, vtb, abf);
    conv(wo, wb, (long long)DD * DD);
    gemm(abf, wb, cattn, bo, NTOK, DD, DD, 1, 0, 0, 0);
    k_add<<<1024, 256, 0, stream>>>(x, cattn, (long long)NTOK * DD);

    // ---- MoE block ----
    k_rmsnorm<<<NTOK, 256, 0, stream>>>(x, ln2, xn);
    k_router<<<NTOK, 32, 0, stream>>>(xn, wr, br, topi, gatev);
    k_capacity<<<EE, 32, 0, stream>>>(topi, gatev, slot, sgate, cnt);
    k_gather<<<dim3(CAP, EE), 256, 0, stream>>>(xn, slot, cnt, xg);
    conv(w1, wb, (long long)EE * HID * DD);
    gemm(xg, wb, c1, nullptr, CAP, HID, DD, EE,
         (long long)CAP * DD, (long long)HID * DD, (long long)CAP * HID);
    conv(w3, wb, (long long)EE * HID * DD);
    gemm(xg, wb, c3, nullptr, CAP, HID, DD, EE,
         (long long)CAP * DD, (long long)HID * DD, (long long)CAP * HID);
    k_silu_mul<<<8192, 256, 0, stream>>>(c1, c3, hb, (long long)EE * CAP * HID);
    conv(w2, wb, (long long)EE * DD * HID);
    gemm(hb, wb, yo, nullptr, CAP, DD, HID, EE,
         (long long)CAP * HID, (long long)DD * HID, (long long)CAP * DD);
    k_scatter<<<dim3(CAP, EE), 256, 0, stream>>>(yo, slot, sgate, cnt, x);
  }

  // ---- final norm + logits ----
  k_rmsnorm<<<NTOK, 256, 0, stream>>>(x, lnf, xn);
  conv(finalw, wb, (long long)VV * DD);
  gemm(xn, wb, (float*)d_out, nullptr, NTOK, VV, DD, 1, 0, 0, 0);
}
